// BnnDilConv_5428838662851
// MI455X (gfx1250) — 8-bit matrix operand order repaired and checked by reading, compiled, not run on hardware
//
#include <hip/hip_runtime.h>
#include <stdint.h>

typedef int v8i __attribute__((ext_vector_type(8)));

#define C_CH   256
#define NB     32
#define HH     56
#define WW     56
#define HP     60
#define WP     60
#define NPIX   (NB * HH * WW)
#define KTOT   2304
#define PTILES (NPIX / 16)
#define PT_PER_BLK 8
#define PBLKS  (PTILES / PT_PER_BLK)

#define OFF_XB   589824u
#define OFF_Y    30081024u
#define OFF_SUM  81461248u

__global__ void pack_x_kernel(const float* __restrict__ x, int8_t* __restrict__ xb) {
  unsigned idx = blockIdx.x * 256u + threadIdx.x;
  int c = idx & 255;
  unsigned t = idx >> 8;
  int w = t % 56;
  int h = (t / 56) % 56;
  int n = t / 3136;
  float v = x[(((size_t)n * 256 + c) * 56 + h) * 56 + w];
  xb[(((size_t)n * HP + h + 2) * WP + (w + 2)) * 256 + c] = (v >= 0.f) ? (int8_t)1 : (int8_t)-1;
}

__global__ void pack_w_kernel(const float* __restrict__ Wt, int8_t* __restrict__ wb) {
  unsigned idx = blockIdx.x * 256u + threadIdx.x;
  int cin  = idx & 255;
  int tap  = (idx >> 8) % 9;
  int cout = idx / (9 * 256);
  float v = Wt[((size_t)cout * 256 + cin) * 9 + tap];
  wb[(size_t)cout * KTOT + tap * 256 + cin] = (v >= 0.f) ? (int8_t)1 : (int8_t)-1;
}

__global__ void __launch_bounds__(256) bconv_kernel(const int8_t* __restrict__ wb,
                                                    const int8_t* __restrict__ xb,
                                                    short* __restrict__ y,
                                                    float* __restrict__ gsum,
                                                    float* __restrict__ gsq) {
  __shared__ float lsum[16];
  __shared__ float lsq[16];

  const int tid  = threadIdx.x;
  const int lane = tid & 31;
  const int wave = tid >> 5;
  const int hh   = lane >> 4;
  const int l15  = lane & 15;

  const int ctile = blockIdx.x / PBLKS;
  const int pblk  = blockIdx.x % PBLKS;
  const int ptile = pblk * PT_PER_BLK + wave;
  const int pixel = ptile * 16 + l15;

  const int img = pixel / 3136;
  const int rem = pixel % 3136;
  const int oh  = rem / 56;
  const int ow  = rem % 56;

  if (tid < 16) { lsum[tid] = 0.f; lsq[tid] = 0.f; }
  __syncthreads();

  const int m = ctile * 16 + l15;
  const int8_t* Abase = wb + (size_t)m * KTOT + hh * 16;
  const int8_t* Bbase = xb + (((size_t)img * HP + oh) * WP + ow) * 256 + hh * 16;

  v8i acc = {0, 0, 0, 0, 0, 0, 0, 0};

  for (int kh = 0; kh < 3; ++kh) {
    for (int kw = 0; kw < 3; ++kw) {
      const int tap = kh * 3 + kw;
      const int8_t* At = Abase + tap * 256;
      const int8_t* Bt = Bbase + ((2 * kh) * WP + 2 * kw) * 256;
      for (int kb = 0; kb < 4; ++kb) {
        union { v8i v; uint2 u2[4]; } a;
        union { v8i v; uint4 u4[2]; } b;
        const int8_t* Ak = At + kb * 64;
        a.u2[0] = *(const uint2*)(Ak + 0);
        a.u2[1] = *(const uint2*)(Ak + 8);
        a.u2[2] = *(const uint2*)(Ak + 32);
        a.u2[3] = *(const uint2*)(Ak + 40);
        const int8_t* Bk = Bt + kb * 64;
        b.u4[0] = *(const uint4*)(Bk + 0);
        b.u4[1] = *(const uint4*)(Bk + 32);
        acc = __builtin_amdgcn_wmma_i32_16x16x64_iu8(true, a.v, true, b.v, acc,
                                                     false, false);
      }
    }
  }

  union { v8i v; int i[8]; } r;
  r.v = acc;
#pragma unroll
  for (int rr = 0; rr < 8; ++rr) {
    const int cval = r.i[rr];
    const int cch  = ctile * 16 + rr + hh * 8;
    y[(size_t)cch * NPIX + pixel] = (short)cval;

    float s = (float)cval;
    float q = s * s;
#pragma unroll
    for (int mk = 1; mk <= 8; mk <<= 1) {
      s += __shfl_xor(s, mk, 32);
      q += __shfl_xor(q, mk, 32);
    }
    if (l15 == 0) {
      atomicAdd(&lsum[rr + hh * 8], s);
      atomicAdd(&lsq[rr + hh * 8], q);
    }
  }
  __syncthreads();
  if (tid < 16) {
    atomicAdd(&gsum[ctile * 16 + tid], lsum[tid]);
    atomicAdd(&gsq[ctile * 16 + tid], lsq[tid]);
  }
}

__global__ void stats_kernel(const float* __restrict__ gsum, const float* __restrict__ gsq,
                             const float* __restrict__ gamma, const float* __restrict__ beta,
                             float* __restrict__ scale, float* __restrict__ bias) {
  int c = threadIdx.x;
  const float cnt = (float)NPIX;
  float mean = gsum[c] / cnt;
  float var  = gsq[c] / cnt - mean * mean;
  float inv  = rsqrtf(var + 1e-5f);
  float g    = gamma[c] * inv;
  scale[c] = g;
  bias[c]  = beta[c] - mean * g;
}

__global__ void finish_kernel(const short* __restrict__ y, const float* __restrict__ scale,
                              const float* __restrict__ bias, float* __restrict__ out) {
  size_t idx = (size_t)blockIdx.x * 256u + threadIdx.x;
  int hw = (int)(idx % 3136);
  int c  = (int)((idx / 3136) % 256);
  int n  = (int)(idx / (3136u * 256u));
  float v = (float)y[(size_t)c * NPIX + (size_t)n * 3136 + hw];
  out[idx] = tanhf(v * scale[c] + bias[c]);
}

extern "C" void kernel_launch(void* const* d_in, const int* in_sizes, int n_in,
                              void* d_out, int out_size, void* d_ws, size_t ws_size,
                              hipStream_t stream) {
  const float* x     = (const float*)d_in[0];
  const float* Wt    = (const float*)d_in[1];
  const float* gamma = (const float*)d_in[2];
  const float* beta  = (const float*)d_in[3];
  float* out = (float*)d_out;

  char* ws = (char*)d_ws;
  int8_t* wb   = (int8_t*)ws;
  int8_t* xb   = (int8_t*)(ws + OFF_XB);
  short*  y    = (short*)(ws + OFF_Y);
  float*  gsum = (float*)(ws + OFF_SUM);
  float*  gsq  = gsum + 256;
  float*  scl  = gsum + 512;
  float*  bia  = gsum + 768;

  hipMemsetAsync(xb, 0, (size_t)NB * HP * WP * 256, stream);
  hipMemsetAsync(gsum, 0, 512 * sizeof(float), stream);

  pack_x_kernel<<<NPIX, 256, 0, stream>>>(x, xb);
  pack_w_kernel<<<2304, 256, 0, stream>>>(Wt, wb);
  bconv_kernel<<<16 * PBLKS, 256, 0, stream>>>(wb, xb, y, gsum, gsq);
  stats_kernel<<<1, 256, 0, stream>>>(gsum, gsq, gamma, beta, scl, bia);
  finish_kernel<<<NPIX, 256, 0, stream>>>(y, scl, bia, out);
}
